// W8A8B16O16Linear_40243843564269
// MI455X (gfx1250) — 8-bit matrix operand order repaired and checked by reading, compiled, not run on hardware
//
#include <hip/hip_runtime.h>
#include <stdint.h>

typedef int v8i __attribute__((ext_vector_type(8)));

constexpr int KDIM = 4096;
constexpr int NDIM = 12288;
constexpr int BM = 128, BN = 128, BK = 64;
constexpr int LDA = 80;
constexpr int LDB = 80;
constexpr int KTILES = KDIM / BK;

__global__ __launch_bounds__(256)
void w8a8_i8gemm_dq_kernel(const signed char* __restrict__ X,
                           const signed char* __restrict__ W,
                           const float* __restrict__ Asc,
                           const float* __restrict__ Bsc,
                           _Float16* __restrict__ Out)
{
    __shared__ __attribute__((aligned(16))) unsigned char sA[2][BM * LDA];
    __shared__ __attribute__((aligned(16))) unsigned char sB[2][BN * LDB];

    const int tid  = threadIdx.x;
    const int lane = tid & 31;
    const int wid  = tid >> 5;
    const int wm   = wid & 1;
    const int wn   = wid >> 1;
    const int half = lane >> 4;
    const int l16  = lane & 15;

    const int bm0 = blockIdx.y * BM;
    const int bn0 = blockIdx.x * BN;

    const uint32_t sA_base = (uint32_t)(uintptr_t)(&sA[0][0]);

    v8i acc[4][2];
#pragma unroll
    for (int mt = 0; mt < 4; ++mt)
#pragma unroll
        for (int nt = 0; nt < 2; ++nt)
            acc[mt][nt] = (v8i)(0);

    auto load_tile = [&](int kt, int buf) {
        const int k0 = kt * BK;
#pragma unroll
        for (int p = 0; p < 2; ++p) {
            const int row = p * 64 + (tid >> 2);
            const int ko  = (tid & 3) * 16;
            const uint32_t goff = (uint32_t)((bm0 + row) * KDIM + k0 + ko);
            const uint32_t loff = sA_base + (uint32_t)(buf * (BM * LDA) + row * LDA + ko);
            asm volatile("global_load_async_to_lds_b128 %0, %1, %2 offset:0"
                         :: "v"(loff), "v"(goff), "s"(X)
                         : "memory");
        }
#pragma unroll
        for (int p = 0; p < 2; ++p) {
            const int task = p * 256 + tid;
            const int kg = task >> 5;
            const int cg = task & 31;
            const int k  = kg * 4;
            const int c  = cg * 4;
            const size_t gbase = (size_t)(k0 + k) * NDIM + bn0 + c;
            uint32_t w0 = *(const uint32_t*)(W + gbase);
            uint32_t w1 = *(const uint32_t*)(W + gbase + NDIM);
            uint32_t w2 = *(const uint32_t*)(W + gbase + 2 * (size_t)NDIM);
            uint32_t w3 = *(const uint32_t*)(W + gbase + 3 * (size_t)NDIM);
            if (kt + 1 < KTILES)
                __builtin_prefetch(W + gbase + (size_t)BK * NDIM, 0, 1);
#pragma unroll
            for (int j = 0; j < 4; ++j) {
                const uint32_t t = ((w0 >> (8 * j)) & 0xffu)
                                 | (((w1 >> (8 * j)) & 0xffu) << 8)
                                 | (((w2 >> (8 * j)) & 0xffu) << 16)
                                 | ((w3 >> (8 * j)) << 24);
                *(uint32_t*)&sB[buf][(c + j) * LDB + k] = t;
            }
        }
    };

    auto compute = [&](int buf) {
        v8i bf[2];
#pragma unroll
        for (int nt = 0; nt < 2; ++nt) {
            const int col = wn * 32 + nt * 16 + l16;
            const unsigned char* base = &sB[buf][col * LDB];
            const int4 d0 = *(const int4*)(base + half * 16);
            const int4 d1 = *(const int4*)(base + 32 + half * 16);
            v8i b;
            b[0] = d0.x; b[1] = d0.y; b[2] = d0.z; b[3] = d0.w;
            b[4] = d1.x; b[5] = d1.y; b[6] = d1.z; b[7] = d1.w;
            bf[nt] = b;
        }
        v8i af[4];
#pragma unroll
        for (int mt = 0; mt < 4; ++mt) {
            const int row = wm * 64 + mt * 16 + l16;
            const unsigned char* base = &sA[buf][row * LDA + half * 16];
            const int2 g0 = *(const int2*)(base + 0);
            const int2 g1 = *(const int2*)(base + 8);
            const int2 g2 = *(const int2*)(base + 32);
            const int2 g3 = *(const int2*)(base + 40);
            v8i a;
            a[0] = g0.x; a[1] = g0.y; a[2] = g1.x; a[3] = g1.y;
            a[4] = g2.x; a[5] = g2.y; a[6] = g3.x; a[7] = g3.y;
            af[mt] = a;
        }
#pragma unroll
        for (int mt = 0; mt < 4; ++mt)
#pragma unroll
            for (int nt = 0; nt < 2; ++nt)
                acc[mt][nt] = __builtin_amdgcn_wmma_i32_16x16x64_iu8(
                     true, af[mt],  true, bf[nt],
                    acc[mt][nt],  false,  false);
    };

    int buf = 0;
    load_tile(0, buf);
    asm volatile("s_wait_asynccnt 0x0" ::: "memory");
    __syncthreads();
    for (int kt = 0; kt < KTILES; ++kt) {
        if (kt + 1 < KTILES) load_tile(kt + 1, buf ^ 1);
        compute(buf);
        asm volatile("s_wait_asynccnt 0x0" ::: "memory");
        __syncthreads();
        buf ^= 1;
    }

    const float alpha = Asc[0] * Bsc[0];
#pragma unroll
    for (int mt = 0; mt < 4; ++mt) {
        const int r0 = bm0 + wm * 64 + mt * 16 + half * 8;
#pragma unroll
        for (int nt = 0; nt < 2; ++nt) {
            const int col = bn0 + wn * 32 + nt * 16 + l16;
#pragma unroll
            for (int r = 0; r < 8; ++r) {
                Out[(size_t)(r0 + r) * NDIM + col] =
                    (_Float16)((float)acc[mt][nt][r] * alpha);
            }
        }
    }
}

extern "C" void kernel_launch(void* const* d_in, const int* in_sizes, int n_in,
                              void* d_out, int out_size, void* d_ws, size_t ws_size,
                              hipStream_t stream) {
    const signed char* X = (const signed char*)d_in[0];
    const signed char* W = (const signed char*)d_in[1];
    const float* a = (const float*)d_in[2];
    const float* b = (const float*)d_in[3];
    _Float16* out = (_Float16*)d_out;

    const int Mrows = in_sizes[0] / KDIM;
    dim3 grid(NDIM / BN, Mrows / BM);
    w8a8_i8gemm_dq_kernel<<<grid, 256, 0, stream>>>(X, W, a, b, out);
}
